// AttnConv2d_25864293056691
// MI455X (gfx1250) — matrix operand order repaired and checked by reading, not compiled, not run on hardware
//
#include <hip/hip_runtime.h>

// ---------------------------------------------------------------------------
// AttnConv2d on MI455X (gfx1250), all-WMMA pipeline, NHWC shift-GEMM convs.
//   conv{1,2,3}: 9 shifted 96x96 GEMMs over NHWC bf16 input. Weight panel
//                double-buffered in LDS (2x18KB, one barrier per shift);
//                each wave processes TWO 16-pixel columns so every A
//                fragment (LDS) feeds 2 WMMAs.
//   attention  : 144 batched 96x96 GEMMs, K=1024; wave computes 16x32 tile.
//   softmax    : rows of 864, scale 1/sqrt(864); emits per-(b,kidx) panels.
//   final      : == 3x3 conv of conv3 (NHWC) with per-batch weight panels.
// ---------------------------------------------------------------------------

typedef __bf16 bf16_t;
typedef __attribute__((ext_vector_type(16))) __bf16 v16bf;
typedef __attribute__((ext_vector_type(8)))  __bf16 v8bf;
typedef __attribute__((ext_vector_type(8)))  float  v8f;

#define B_   16
#define C_   96
#define H_   96
#define W_   96
#define HP   98
#define WP   98
#define K9   864          // C_*9
#define HW_  9216         // H_*W_
#define L_   1024         // (96/3)^2
#define NTOT (B_*HW_)     // 147456
#define CC_  (C_*C_)      // 9216: one weight panel (18KB bf16)

union AFrag { v16bf v; v8bf h[2]; };

__device__ __forceinline__ v8f wmma_bf16(v16bf a, v16bf b, v8f c) {
  // (neg_a, A, neg_b, B, c_mod, C, reuse_a, reuse_b)
  return __builtin_amdgcn_wmma_f32_16x16x32_bf16(false, a, false, b,
                                                 (short)0, c, false, false);
}

// ------------------------------------------------- pad + cast to NHWC bf16
// xp is pre-zeroed (memset); only interior written. Coalesced read of NCHW.
__global__ __launch_bounds__(256) void pad_cast_nhwc_kernel(
    const float* __restrict__ x, bf16_t* __restrict__ xp) {
  long idx = (long)blockIdx.x * blockDim.x + threadIdx.x;  // over B*C*H*W
  if (idx >= (long)B_ * C_ * H_ * W_) return;
  int w = (int)(idx % W_); long t = idx / W_;
  int h = (int)(t % H_);   t /= H_;
  int c = (int)(t % C_);   int b = (int)(t / C_);
  xp[(((long)b * HP + (h + 1)) * WP + (w + 1)) * C_ + c] = (bf16_t)x[idx];
}

// Weights: OIHW fp32 -> per-shift panels wpk[kidx][co][ci] bf16.
__global__ __launch_bounds__(256) void cast_w_kernel(
    const float* __restrict__ in, bf16_t* __restrict__ out) {
  int idx = blockIdx.x * blockDim.x + threadIdx.x;  // over 9*96*96
  if (idx >= 9 * CC_) return;
  int ci = idx % C_;
  int co = (idx / C_) % C_;
  int r  = idx / CC_;                       // kh*3+kw
  out[idx] = (bf16_t)in[(co * C_ + ci) * 9 + r];
}

// ---- store one 96(M) x 16(N) result column -------------------------------
__device__ __forceinline__ void store_tile(
    const v8f* acc, int mode, int b, int h, int w, int hw, int hi,
    bf16_t* __restrict__ out_bf, float* __restrict__ out_f) {
  if (mode <= 1) {            // unfold(non-overlap) layout [b][kidx][c][l]
    const int kidx = (h % 3) * 3 + (w % 3);
    const int l    = (h / 3) * 32 + (w / 3);
    bf16_t* y = out_bf + ((long)(b * 9 + kidx)) * C_ * L_ + l;
#pragma unroll
    for (int m = 0; m < 6; ++m)
#pragma unroll
      for (int r = 0; r < 8; ++r) {
        int co = m * 16 + hi * 8 + r;
        y[(long)co * L_] = (bf16_t)acc[m][r];
      }
  } else if (mode == 2) {     // NHWC padded bf16, packed 16B stores
    bf16_t* yb = out_bf + (((long)b * HP + (h + 1)) * WP + (w + 1)) * C_;
#pragma unroll
    for (int m = 0; m < 6; ++m) {
      v8bf pk;
#pragma unroll
      for (int r = 0; r < 8; ++r) pk[r] = (bf16_t)acc[m][r];
      *(v8bf*)(yb + m * 16 + hi * 8) = pk;
    }
  } else {                    // fp32 NCHW output
    float* yo = out_f + ((long)b * C_) * HW_ + hw;
#pragma unroll
    for (int m = 0; m < 6; ++m)
#pragma unroll
      for (int r = 0; r < 8; ++r) {
        int co = m * 16 + hi * 8 + r;
        yo[(long)co * HW_] = acc[m][r];
      }
  }
}

// ------------------------------------------------- shift-GEMM 3x3 conv
// Workgroup: 8 waves, tile M=96 x N=256 pixels (each wave: two 16-pixel
// columns). K = 9 shifts x 96 channels. Weight panels double-buffered in
// LDS: next panel's global loads issue before compute, land in the
// alternate buffer after compute; one barrier per shift.
__global__ __launch_bounds__(256) void conv_gemm_kernel(
    const bf16_t* __restrict__ xp, const bf16_t* __restrict__ wts,
    int per_batch, int mode,
    bf16_t* __restrict__ out_bf, float* __restrict__ out_f) {
  __shared__ __align__(16) bf16_t lds_w[2][CC_];   // 2 x 18KB A panels

  const int tid  = threadIdx.x;
  const int wave = tid >> 5, lane = tid & 31;
  const int l16  = lane & 15, hi = lane >> 4;

  const int p0 = blockIdx.x * 256 + wave * 16 + l16;  // pixel, tile 0
  const int p1 = p0 + 128;                            // pixel, tile 1
  const int b  = p0 / HW_;                            // same image for both
  const int hw0 = p0 % HW_, hw1 = p1 % HW_;
  const int h0 = hw0 / W_, w0 = hw0 % W_;
  const int h1 = hw1 / W_, w1 = hw1 % W_;

  const bf16_t* wb = wts + (per_batch ? (long)b * 9 * CC_ : 0);
  const uint2*  wsrc = (const uint2*)wb;              // 2304 uint2 per panel
  const bf16_t* xpix0 = xp + (((long)b * HP + h0) * WP + w0) * C_;
  const bf16_t* xpix1 = xp + (((long)b * HP + h1) * WP + w1) * C_;

  // prologue: stage panel 0 into buffer 0
  {
    uint2* dst = (uint2*)lds_w[0];
#pragma unroll
    for (int i = 0; i < 9; ++i) dst[tid + i * 256] = wsrc[tid + i * 256];
  }
  __syncthreads();

  v8f zero = {};
  v8f acc0[6], acc1[6];
#pragma unroll
  for (int m = 0; m < 6; ++m) { acc0[m] = zero; acc1[m] = zero; }

  for (int kidx = 0; kidx < 9; ++kidx) {
    // ---- issue next panel's global loads (latency hidden by compute)
    uint2 stage[9];
    if (kidx < 8) {
      const uint2* src = wsrc + (size_t)(kidx + 1) * (CC_ / 4);
#pragma unroll
      for (int i = 0; i < 9; ++i) stage[i] = src[tid + i * 256];
    }

    const bf16_t* lw = lds_w[kidx & 1];
    const int kh = kidx / 3, kw = kidx - (kidx / 3) * 3;
    const int shift = (kh * WP + kw) * C_;
    const bf16_t* brow0 = xpix0 + shift + hi * 16;
    const bf16_t* brow1 = xpix1 + shift + hi * 16;
    if (kidx < 8) {  // prefetch next shift's B rows
      int nk = kidx + 1;
      int ns = ((nk / 3) * WP + (nk % 3)) * C_;
      __builtin_prefetch(xpix0 + ns, 0, 3);
      __builtin_prefetch(xpix1 + ns, 0, 3);
    }
#pragma unroll
    for (int kc = 0; kc < 3; ++kc) {
      v16bf bv0 = *(const v16bf*)(brow0 + kc * 32);   // 32B aligned channels
      v16bf bv1 = *(const v16bf*)(brow1 + kc * 32);
      AFrag af[6];
#pragma unroll
      for (int m = 0; m < 6; ++m) {
        const bf16_t* arow = lw + (m * 16 + l16) * C_ + kc * 32;
        af[m].h[0] = *(const v8bf*)(arow + hi * 16);
        af[m].h[1] = *(const v8bf*)(arow + hi * 16 + 8);
      }
#pragma unroll
      for (int m = 0; m < 6; ++m) acc0[m] = wmma_bf16(af[m].v, bv0, acc0[m]);
#pragma unroll
      for (int m = 0; m < 6; ++m) acc1[m] = wmma_bf16(af[m].v, bv1, acc1[m]);
    }

    // ---- park next panel in the alternate LDS buffer; single barrier
    if (kidx < 8) {
      uint2* dst = (uint2*)lds_w[(kidx + 1) & 1];
#pragma unroll
      for (int i = 0; i < 9; ++i) dst[tid + i * 256] = stage[i];
    }
    __syncthreads();
  }

  store_tile(acc0, mode, b, h0, w0, hw0, hi, out_bf, out_f);
  store_tile(acc1, mode, b, h1, w1, hw1, hi, out_bf, out_f);
}

// ------------------------------------------ attention GEMM: Y1 * Y2^T
// One wave per 16x32 tile (A fragment reused for two B columns).
// grid = (18 = 6 mt x 3 ntPair, 9 kidx, 16 batch), K = 1024.
__global__ __launch_bounds__(32) void attn_gemm_kernel(
    const bf16_t* __restrict__ y1u, const bf16_t* __restrict__ y2u,
    float* __restrict__ attn_f) {
  const int lane = threadIdx.x & 31;
  const int l16  = lane & 15, hi = lane >> 4;
  const int mt = blockIdx.x / 3, nt0 = (blockIdx.x % 3) * 2;
  const int kidx = blockIdx.y, b = blockIdx.z;

  const bf16_t* Y1 = y1u + (long)(b * 9 + kidx) * C_ * L_;
  const bf16_t* Y2 = y2u + (long)(b * 9 + kidx) * C_ * L_;
  const bf16_t* a_row  = Y1 + (long)(mt * 16 + l16) * L_;         // A: [c][l]
  const bf16_t* b_row0 = Y2 + (long)(nt0 * 16 + l16) * L_;        // col d
  const bf16_t* b_row1 = Y2 + (long)((nt0 + 1) * 16 + l16) * L_;

  v8f acc0 = {}, acc1 = {};
  for (int lc = 0; lc < 32; ++lc) {
    AFrag af;
    const bf16_t* ap = a_row + lc * 32;
    af.h[0] = *(const v8bf*)(ap + hi * 16);
    af.h[1] = *(const v8bf*)(ap + hi * 16 + 8);
    v16bf bv0 = *(const v16bf*)(b_row0 + lc * 32 + hi * 16);
    v16bf bv1 = *(const v16bf*)(b_row1 + lc * 32 + hi * 16);
    acc0 = wmma_bf16(af.v, bv0, acc0);
    acc1 = wmma_bf16(af.v, bv1, acc1);
  }
  // attn2[b, d, c*9 + kidx] = attn[b, kidx, c, d]
#pragma unroll
  for (int r = 0; r < 8; ++r) {
    int c = mt * 16 + hi * 8 + r;
    long base = ((long)b * C_) * K9 + c * 9 + kidx;
    attn_f[base + (long)(nt0 * 16 + l16) * K9]       = acc0[r];
    attn_f[base + (long)((nt0 + 1) * 16 + l16) * K9] = acc1[r];
  }
}

// --------------------------------------------------------------- softmax
// Row (b,d) of 864; output scattered into per-(b,kidx) panels [d][c].
__global__ __launch_bounds__(256) void softmax_kernel(
    const float* __restrict__ attn_f, bf16_t* __restrict__ a2p) {
  __shared__ float red[256];
  const int row = blockIdx.x;  // b*96 + d
  const int b = row / C_, d = row % C_;
  const int tid = threadIdx.x;
  const float invT = 0.03402069087198862f;  // 1/sqrt(864)
  const float* src = attn_f + (long)row * K9;

  float mx = -3.4e38f;
  for (int i = tid; i < K9; i += 256) mx = fmaxf(mx, src[i] * invT);
  red[tid] = mx; __syncthreads();
  for (int s = 128; s > 0; s >>= 1) {
    if (tid < s) red[tid] = fmaxf(red[tid], red[tid + s]);
    __syncthreads();
  }
  mx = red[0]; __syncthreads();

  float sum = 0.f;
  for (int i = tid; i < K9; i += 256) sum += __expf(src[i] * invT - mx);
  red[tid] = sum; __syncthreads();
  for (int s = 128; s > 0; s >>= 1) {
    if (tid < s) red[tid] += red[tid + s];
    __syncthreads();
  }
  const float inv = 1.0f / red[0];

  // m = c*9 + kidx  ->  panel a2p[b][kidx][d][c]
  for (int i = tid; i < K9; i += 256) {
    int c = i / 9, kidx = i % 9;
    float v = __expf(src[i] * invT - mx) * inv;
    a2p[(((long)b * 9 + kidx) * C_ + d) * C_ + c] = (bf16_t)v;
  }
}

// ---------------------------------------------------------------------------
extern "C" void kernel_launch(void* const* d_in, const int* in_sizes, int n_in,
                              void* d_out, int out_size, void* d_ws, size_t ws_size,
                              hipStream_t stream) {
  const float* x  = (const float*)d_in[0];
  const float* W1 = (const float*)d_in[1];
  const float* W2 = (const float*)d_in[2];
  const float* W3 = (const float*)d_in[3];

  char* ws = (char*)d_ws;
  size_t off = 0;
  const size_t sz_pad  = (size_t)B_ * HP * WP * C_ * sizeof(bf16_t); // 29.5 MB
  const size_t sz_unf  = (size_t)B_ * 9 * C_ * L_ * sizeof(bf16_t);  // 28.3 MB
  const size_t sz_w    = (size_t)9 * CC_ * sizeof(bf16_t);           // 166 KB
  const size_t sz_attn = (size_t)B_ * C_ * K9 * sizeof(float);       // 5.3 MB
  const size_t sz_a2p  = (size_t)B_ * 9 * CC_ * sizeof(bf16_t);      // 2.7 MB

  bf16_t* xp    = (bf16_t*)(ws + off); off += sz_pad;   // NHWC padded input
  bf16_t* c3p   = (bf16_t*)(ws + off); off += sz_pad;   // NHWC padded conv3
  bf16_t* y1u   = (bf16_t*)(ws + off); off += sz_unf;
  bf16_t* y2u   = (bf16_t*)(ws + off); off += sz_unf;
  bf16_t* w1p   = (bf16_t*)(ws + off); off += sz_w;
  bf16_t* w2p   = (bf16_t*)(ws + off); off += sz_w;
  bf16_t* w3p   = (bf16_t*)(ws + off); off += sz_w;
  float*  attnf = (float*) (ws + off); off += sz_attn;
  bf16_t* a2p   = (bf16_t*)(ws + off); off += sz_a2p;   // [b][kidx][d][c]

  // 1) zero padded buffers; cast x -> NHWC bf16; weights -> shift panels
  hipMemsetAsync(xp,  0, sz_pad, stream);
  hipMemsetAsync(c3p, 0, sz_pad, stream);
  long nin = (long)B_ * C_ * H_ * W_;
  pad_cast_nhwc_kernel<<<(unsigned)((nin + 255) / 256), 256, 0, stream>>>(x, xp);
  int nw = 9 * CC_;
  cast_w_kernel<<<(nw + 255) / 256, 256, 0, stream>>>(W1, w1p);
  cast_w_kernel<<<(nw + 255) / 256, 256, 0, stream>>>(W2, w2p);
  cast_w_kernel<<<(nw + 255) / 256, 256, 0, stream>>>(W3, w3p);

  // 2) three shift-GEMM convs (WMMA bf16, double-buffered LDS panels)
  dim3 cg(NTOT / 256);
  conv_gemm_kernel<<<cg, 256, 0, stream>>>(xp, w1p, 0, 0, y1u, nullptr);
  conv_gemm_kernel<<<cg, 256, 0, stream>>>(xp, w2p, 0, 1, y2u, nullptr);
  conv_gemm_kernel<<<cg, 256, 0, stream>>>(xp, w3p, 0, 2, c3p, nullptr);

  // 3) batched attention GEMM + softmax (-> per-batch weight panels)
  attn_gemm_kernel<<<dim3(18, 9, 16), 32, 0, stream>>>(y1u, y2u, attnf);
  softmax_kernel<<<B_ * C_, 256, 0, stream>>>(attnf, a2p);

  // 4) final einsum == 3x3 conv of conv3 (NHWC) with per-batch panels
  conv_gemm_kernel<<<cg, 256, 0, stream>>>(c3p, a2p, 1, 3, nullptr, (float*)d_out);
}
